// StateSpaceEncoder_40870908789228
// MI455X (gfx1250) — matrix operand order repaired and checked by reading, not compiled, not run on hardware
//
#include <hip/hip_runtime.h>
#include <hip/hip_bf16.h>

// ---------------------------------------------------------------------------
// MI455X (gfx1250) SSM encoder.
// GEMMs: v_wmma_f32_16x16x32_f16, A-operand staged through LDS with
// global_load_async_to_lds_b128 (ASYNCcnt) double-buffering, B-operand
// register double-buffered from global. Wave coordinates are scalarized with
// readfirstlane so tile predicates are SALU branches (no EXEC toggling around
// WMMA). Recurrence = 3-pass chunked scan.
// ---------------------------------------------------------------------------

typedef __attribute__((ext_vector_type(16))) _Float16 v16h;
typedef __attribute__((ext_vector_type(8)))  _Float16 v8h;
typedef __attribute__((ext_vector_type(8)))  float    v8f;

#define BN   4
#define TT   2048
#define DM   1024
#define SS   64
#define LL   4
#define FF   4096
#define MROWS (BN * TT)
#define NCHUNK 64
#define CHUNKT 32

// LDS A-slab: 128 rows x 32 halfs, row stride padded to 40 halfs (80B) so the
// 16-row ds_load_b128 fragment reads hit distinct bank groups.
#define A_STRIDE_H 40
#define A_SLAB_H  (128 * A_STRIDE_H)

// ---------------- WMMA fragment helpers (ISA 7.12.2 layouts) ----------------

// B fragment: 32x16 f16 tile; weights stored transposed [N x K] row-major.
// lanes 0-15: col N=lane, elems 0..15 = K 0..7,16..23 ; lanes 16-31: K 8..15,24..31
__device__ inline v16h load_frag_b(const _Float16* __restrict__ Bt, int ldb,
                                   int col0, int k0, int lane) {
  const int n  = lane & 15;
  const int kl = (lane >> 4) << 3;  // 0 or 8
  const _Float16* p = Bt + (size_t)(col0 + n) * ldb + k0 + kl;
  v8h lo = *(const v8h*)(p);
  v8h hi = *(const v8h*)(p + 16);
  return __builtin_shufflevector(lo, hi, 0,1,2,3,4,5,6,7,8,9,10,11,12,13,14,15);
}

__device__ inline float gelu_exact(float v) {
  return 0.5f * v * (1.0f + erff(v * 0.70710678118654752f));
}

// ---------------- WMMA GEMM -------------------------------------------------
// C[M,N] = A[M,K] * Bt[N,K]^T. Block 256 threads = 8 waves (2 x 4).
// Block tile 128x128; wave tile 64x32 (4x2 WMMA 16x16 accumulators).
// A slab (128x32 f16) staged to LDS via async-to-LDS, double buffered.
// EPI: 0 = store f32 ; 1 = f32 += ; 2 = f16 gelu(acc+bias) ; 3 = f32 += acc+bias
template<int EPI>
__global__ __launch_bounds__(256) void gemm_wmma_kernel(
    const _Float16* __restrict__ A, const _Float16* __restrict__ Bt,
    const float* __restrict__ bias, float* __restrict__ Of,
    _Float16* __restrict__ Oh, int M, int N, int K) {
  __shared__ __attribute__((aligned(16))) _Float16 sA[2 * A_SLAB_H];

  const int tid  = threadIdx.x;
  const int lane = tid & 31;
  // wave id is wave-uniform: pin it to an SGPR so all tile coordinates and
  // predicates below are scalar (WMMA requires EXEC all-1s; avoid saveexec).
  const int wave = __builtin_amdgcn_readfirstlane(tid >> 5);
  const int wm = wave >> 2;                       // 0..1   (scalar)
  const int wn = wave & 3;                        // 0..3   (scalar)
  const int rowBase = blockIdx.y * 128;
  const int row0 = rowBase + wm * 64;             // scalar
  const int col0 = blockIdx.x * 128 + wn * 32;    // scalar
  const bool active = (col0 < N);                 // scalar

  // LDS byte offset of sA (generic-pointer low 32 bits == LDS offset)
  const unsigned sA_base = (unsigned)(unsigned long long)(uintptr_t)&sA[0];

  // stage one 128x32 A slab into LDS buffer `buf` with async-to-LDS copies;
  // 2 passes x 256 threads x 16B = 8KB of payload.
  auto stage_a = [&](int buf, int k0) {
#pragma unroll
    for (int p = 0; p < 2; ++p) {
      const int e   = tid + p * 256;
      const int row = e >> 2;                     // 0..127
      const int q   = e & 3;                      // 16B quarter of the 64B row
      const unsigned loff =
          sA_base + (unsigned)(buf * A_SLAB_H * 2 + row * (A_STRIDE_H * 2) + q * 16);
      const unsigned long long gaddr =
          (unsigned long long)(uintptr_t)(A + (size_t)(rowBase + row) * K + k0 + q * 8);
      asm volatile("global_load_async_to_lds_b128 %0, %1, off"
                   :: "v"(loff), "v"(gaddr) : "memory");
    }
  };

  // A fragment read from LDS (ds_load_b128 x2):
  // lanes 0-15: row = rowInSlab+lane, kh=0 ; lanes 16-31: same rows, kh=8
  auto lds_frag_a = [&](int buf, int rowInSlab) -> v16h {
    const int r  = rowInSlab + (lane & 15);
    const int kh = (lane >> 4) << 3;              // 0 or 8 (halfs)
    const _Float16* p = &sA[buf * A_SLAB_H + r * A_STRIDE_H + kh];
    v8h lo = *(const v8h*)(p);
    v8h hi = *(const v8h*)(p + 16);
    return __builtin_shufflevector(lo, hi,
                                   0,1,2,3,4,5,6,7,8,9,10,11,12,13,14,15);
  };

  v8f acc[4][2];
#pragma unroll
  for (int i = 0; i < 4; ++i)
#pragma unroll
    for (int j = 0; j < 2; ++j)
      acc[i][j] = (v8f){0.f, 0.f, 0.f, 0.f, 0.f, 0.f, 0.f, 0.f};

  // prologue: stage slab 0, preload B frags for k0 = 0
  int buf = 0;
  stage_a(0, 0);
  v16h bcur0, bcur1;
  if (active) {
    bcur0 = load_frag_b(Bt, K, col0, 0, lane);
    bcur1 = load_frag_b(Bt, K, col0 + 16, 0, lane);
  }
  asm volatile("s_wait_asynccnt 0x0" ::: "memory");
  __syncthreads();

  for (int k0 = 0; k0 < K; k0 += 32) {
    const bool notlast = (k0 + 32) < K;           // block-uniform scalar
    if (notlast) stage_a(buf ^ 1, k0 + 32);       // overlap with WMMAs below

    v16h bnext0, bnext1;
    if (notlast && active) {
      __builtin_prefetch((const void*)(Bt + (size_t)(col0 + (lane & 15)) * K + k0 + 64), 0, 3);
      bnext0 = load_frag_b(Bt, K, col0, k0 + 32, lane);
      bnext1 = load_frag_b(Bt, K, col0 + 16, k0 + 32, lane);
    }

    if (active) {
      // batch the four DS fragment loads, then run the 8-WMMA burst
      v16h a0 = lds_frag_a(buf, wm * 64);
      v16h a1 = lds_frag_a(buf, wm * 64 + 16);
      v16h a2 = lds_frag_a(buf, wm * 64 + 32);
      v16h a3 = lds_frag_a(buf, wm * 64 + 48);
      acc[0][0] = __builtin_amdgcn_wmma_f32_16x16x32_f16(false, a0, false, bcur0, (short)0, acc[0][0], false, false);
      acc[0][1] = __builtin_amdgcn_wmma_f32_16x16x32_f16(false, a0, false, bcur1, (short)0, acc[0][1], false, false);
      acc[1][0] = __builtin_amdgcn_wmma_f32_16x16x32_f16(false, a1, false, bcur0, (short)0, acc[1][0], false, false);
      acc[1][1] = __builtin_amdgcn_wmma_f32_16x16x32_f16(false, a1, false, bcur1, (short)0, acc[1][1], false, false);
      acc[2][0] = __builtin_amdgcn_wmma_f32_16x16x32_f16(false, a2, false, bcur0, (short)0, acc[2][0], false, false);
      acc[2][1] = __builtin_amdgcn_wmma_f32_16x16x32_f16(false, a2, false, bcur1, (short)0, acc[2][1], false, false);
      acc[3][0] = __builtin_amdgcn_wmma_f32_16x16x32_f16(false, a3, false, bcur0, (short)0, acc[3][0], false, false);
      acc[3][1] = __builtin_amdgcn_wmma_f32_16x16x32_f16(false, a3, false, bcur1, (short)0, acc[3][1], false, false);
    }

    if (notlast) {
      asm volatile("s_wait_asynccnt 0x0" ::: "memory");
      __syncthreads();
      buf ^= 1;
      bcur0 = bnext0;
      bcur1 = bnext1;
    }
  }

  if (!active) return;

  // epilogue: acc element r holds rows {r, r+8}; lane&15 = column
  const int n  = lane & 15;
  const int mo = (lane >> 4) << 3;
#pragma unroll
  for (int i = 0; i < 4; ++i) {
#pragma unroll
    for (int j = 0; j < 2; ++j) {
      const int c0 = col0 + j * 16;
      const float bv = (EPI == 2 || EPI == 3) ? bias[c0 + n] : 0.f;
#pragma unroll
      for (int r = 0; r < 8; ++r) {
        const size_t idx = (size_t)(row0 + i * 16 + mo + r) * N + c0 + n;
        float v = acc[i][j][r];
        if (EPI == 0)      Of[idx] = v;
        else if (EPI == 1) Of[idx] += v;
        else if (EPI == 2) Oh[idx] = (_Float16)gelu_exact(v + bv);
        else               Of[idx] += v + bv;
      }
    }
  }
}

// ---------------- LayerNorm -------------------------------------------------
__global__ __launch_bounds__(256) void ln_kernel(
    const float* __restrict__ x, const float* __restrict__ gamma,
    const float* __restrict__ beta, _Float16* __restrict__ oh,
    float* __restrict__ of, int D) {
  const int row = blockIdx.x;
  const float* xr = x + (size_t)row * D;
  float ls = 0.f, lq = 0.f;
  for (int c = threadIdx.x; c < D; c += 256) {
    float v = xr[c];
    ls += v; lq += v * v;
  }
  __shared__ float s1[256], s2[256];
  s1[threadIdx.x] = ls; s2[threadIdx.x] = lq;
  __syncthreads();
  for (int off = 128; off > 0; off >>= 1) {
    if (threadIdx.x < off) {
      s1[threadIdx.x] += s1[threadIdx.x + off];
      s2[threadIdx.x] += s2[threadIdx.x + off];
    }
    __syncthreads();
  }
  const float mean = s1[0] / (float)D;
  const float var  = s2[0] / (float)D - mean * mean;
  const float inv  = rsqrtf(var + 1e-5f);
  for (int c = threadIdx.x; c < D; c += 256) {
    float v = (xr[c] - mean) * inv * gamma[c] + beta[c];
    if (oh) oh[(size_t)row * D + c] = (_Float16)v;
    else    of[(size_t)row * D + c] = v;
  }
}

// ---------------- Embedding -------------------------------------------------
__global__ __launch_bounds__(256) void embed_kernel(
    const int* __restrict__ tokens, const float* __restrict__ emb,
    const float* __restrict__ pos, float* __restrict__ x, int T, int D) {
  const int row = blockIdx.x;             // 0..MROWS-1
  const int t   = row % T;
  const int tok = tokens[row];
  const float* er = emb + (size_t)tok * D;
  const float* pr = pos + (size_t)t * D;
  float* xr = x + (size_t)row * D;
  for (int c = threadIdx.x; c < D; c += 256) xr[c] = er[c] + pr[c];
}

// ---------------- Weight convert + transpose (fp32 [R][C] -> f16 [C][R]) ----
__global__ __launch_bounds__(256) void transpose_to_f16_kernel(
    const float* __restrict__ src, _Float16* __restrict__ dst, int R, int C) {
  size_t i = (size_t)blockIdx.x * 256 + threadIdx.x;
  if (i >= (size_t)R * C) return;
  int r = (int)(i / C), c = (int)(i % C);
  dst[(size_t)c * R + r] = (_Float16)src[i];
}

// ---------------- Chunked scan: h_t = sigmoid(a)*h_{t-1} + u_t --------------
__device__ inline float sigmoidf_(float v) { return 1.f / (1.f + __expf(-v)); }

__global__ __launch_bounds__(SS) void scan_local_kernel(
    float* __restrict__ u, const float* __restrict__ Adiag,
    float* __restrict__ carry) {
  const int b = blockIdx.x / NCHUNK, c = blockIdx.x % NCHUNK, s = threadIdx.x;
  const float a = sigmoidf_(Adiag[s]);
  float h = 0.f;
  for (int i = 0; i < CHUNKT; ++i) {
    const size_t t = (size_t)c * CHUNKT + i;
    const size_t idx = ((size_t)b * TT + t) * SS + s;
    h = a * h + u[idx];
    u[idx] = h;                       // chunk-local scan value
  }
  carry[((size_t)b * NCHUNK + c) * SS + s] = h;
}

__global__ __launch_bounds__(SS) void scan_carry_kernel(
    const float* __restrict__ carry, const float* __restrict__ Adiag,
    float* __restrict__ hin) {
  const int b = blockIdx.x, s = threadIdx.x;
  const float a = sigmoidf_(Adiag[s]);
  float aCT = a;
#pragma unroll
  for (int j = 0; j < 5; ++j) aCT *= aCT;   // a^32
  float pre = 0.f;
  for (int c = 0; c < NCHUNK; ++c) {
    const size_t idx = ((size_t)b * NCHUNK + c) * SS + s;
    hin[idx] = pre;
    pre = aCT * pre + carry[idx];
  }
}

__global__ __launch_bounds__(SS) void scan_fix_kernel(
    const float* __restrict__ u, const float* __restrict__ Adiag,
    const float* __restrict__ hin, _Float16* __restrict__ Hh) {
  const int b = blockIdx.x / NCHUNK, c = blockIdx.x % NCHUNK, s = threadIdx.x;
  const float a = sigmoidf_(Adiag[s]);
  const float h0 = hin[((size_t)b * NCHUNK + c) * SS + s];
  float pw = a;
  for (int i = 0; i < CHUNKT; ++i) {
    const size_t t = (size_t)c * CHUNKT + i;
    const size_t idx = ((size_t)b * TT + t) * SS + s;
    Hh[idx] = (_Float16)(u[idx] + pw * h0);
    pw *= a;
  }
}

// ---------------------------------------------------------------------------

extern "C" void kernel_launch(void* const* d_in, const int* in_sizes, int n_in,
                              void* d_out, int out_size, void* d_ws, size_t ws_size,
                              hipStream_t stream) {
  const int*   tokens = (const int*)  d_in[0];
  const float* emb    = (const float*)d_in[1];
  const float* pos    = (const float*)d_in[2];
  const float* Adiag  = (const float*)d_in[3];
  const float* Bw     = (const float*)d_in[4];   // (L, DM, SS)
  const float* Cw     = (const float*)d_in[5];   // (L, SS, DM)
  const float* Dw     = (const float*)d_in[6];   // (L, DM, DM)
  const float* w1     = (const float*)d_in[7];   // (L, DM, FF)
  const float* b1     = (const float*)d_in[8];   // (L, FF)
  const float* w2     = (const float*)d_in[9];   // (L, FF, DM)
  const float* b2     = (const float*)d_in[10];  // (L, DM)
  const float* ln1s   = (const float*)d_in[11];
  const float* ln1b   = (const float*)d_in[12];
  const float* ln2s   = (const float*)d_in[13];
  const float* ln2b   = (const float*)d_in[14];
  const float* lnfs   = (const float*)d_in[15];
  const float* lnfb   = (const float*)d_in[16];

  // ---- workspace layout (256B aligned slabs) ----
  char* base = (char*)d_ws;
  size_t off = 0;
  auto alloc = [&](size_t bytes) {
    size_t o = off;
    off = (off + bytes + 255) & ~(size_t)255;
    return (void*)(base + o);
  };
  float*    x    = (float*)   alloc((size_t)MROWS * DM * 4);
  _Float16* xnh  = (_Float16*)alloc((size_t)MROWS * DM * 2);
  float*    u    = (float*)   alloc((size_t)MROWS * SS * 4);
  _Float16* Hh   = (_Float16*)alloc((size_t)MROWS * SS * 2);
  float*    carry= (float*)   alloc((size_t)BN * NCHUNK * SS * 4);
  float*    hin  = (float*)   alloc((size_t)BN * NCHUNK * SS * 4);
  _Float16* gh   = (_Float16*)alloc((size_t)MROWS * FF * 2);
  _Float16* Bt   = (_Float16*)alloc((size_t)LL * SS * DM * 2);   // [l][SS][DM]
  _Float16* Ct   = (_Float16*)alloc((size_t)LL * DM * SS * 2);   // [l][DM][SS]
  _Float16* Dt   = (_Float16*)alloc((size_t)LL * DM * DM * 2);   // [l][DM][DM]
  _Float16* w1t  = (_Float16*)alloc((size_t)LL * FF * DM * 2);   // [l][FF][DM]
  _Float16* w2t  = (_Float16*)alloc((size_t)LL * DM * FF * 2);   // [l][DM][FF]

  auto tgrid = [](size_t n) { return dim3((unsigned)((n + 255) / 256)); };

  // ---- convert + transpose all weights to f16 [N][K] ----
  for (int l = 0; l < LL; ++l) {
    transpose_to_f16_kernel<<<tgrid((size_t)DM * SS), 256, 0, stream>>>(
        Bw + (size_t)l * DM * SS, Bt + (size_t)l * SS * DM, DM, SS);
    transpose_to_f16_kernel<<<tgrid((size_t)SS * DM), 256, 0, stream>>>(
        Cw + (size_t)l * SS * DM, Ct + (size_t)l * DM * SS, SS, DM);
    transpose_to_f16_kernel<<<tgrid((size_t)DM * DM), 256, 0, stream>>>(
        Dw + (size_t)l * DM * DM, Dt + (size_t)l * DM * DM, DM, DM);
    transpose_to_f16_kernel<<<tgrid((size_t)DM * FF), 256, 0, stream>>>(
        w1 + (size_t)l * DM * FF, w1t + (size_t)l * FF * DM, DM, FF);
    transpose_to_f16_kernel<<<tgrid((size_t)FF * DM), 256, 0, stream>>>(
        w2 + (size_t)l * FF * DM, w2t + (size_t)l * DM * FF, FF, DM);
  }

  // ---- embedding ----
  embed_kernel<<<dim3(MROWS), 256, 0, stream>>>(tokens, emb, pos, x, TT, DM);

  auto gemm = [&](int epi, const _Float16* A, const _Float16* Bmat,
                  const float* bias, float* Of, _Float16* Oh,
                  int M_, int N_, int K_) {
    dim3 g((N_ + 127) / 128, M_ / 128), blk(256);
    switch (epi) {
      case 0: gemm_wmma_kernel<0><<<g, blk, 0, stream>>>(A, Bmat, bias, Of, Oh, M_, N_, K_); break;
      case 1: gemm_wmma_kernel<1><<<g, blk, 0, stream>>>(A, Bmat, bias, Of, Oh, M_, N_, K_); break;
      case 2: gemm_wmma_kernel<2><<<g, blk, 0, stream>>>(A, Bmat, bias, Of, Oh, M_, N_, K_); break;
      default: gemm_wmma_kernel<3><<<g, blk, 0, stream>>>(A, Bmat, bias, Of, Oh, M_, N_, K_); break;
    }
  };

  for (int l = 0; l < LL; ++l) {
    const float* Al = Adiag + (size_t)l * SS;
    // LN1 -> xn (f16)
    ln_kernel<<<dim3(MROWS), 256, 0, stream>>>(
        x, ln1s + (size_t)l * DM, ln1b + (size_t)l * DM, xnh, nullptr, DM);
    // u = xn @ B   (M=8192, N=64, K=1024)
    gemm(0, xnh, Bt + (size_t)l * SS * DM, nullptr, u, nullptr, MROWS, SS, DM);
    // chunked diagonal scan -> Hh (f16)
    scan_local_kernel<<<dim3(BN * NCHUNK), SS, 0, stream>>>(u, Al, carry);
    scan_carry_kernel<<<dim3(BN), SS, 0, stream>>>(carry, Al, hin);
    scan_fix_kernel  <<<dim3(BN * NCHUNK), SS, 0, stream>>>(u, Al, hin, Hh);
    // x += H @ C   (K=64, N=1024)
    gemm(1, Hh, Ct + (size_t)l * DM * SS, nullptr, x, nullptr, MROWS, DM, SS);
    // x += xn @ D  (K=1024, N=1024)
    gemm(1, xnh, Dt + (size_t)l * DM * DM, nullptr, x, nullptr, MROWS, DM, DM);
    // LN2 -> xn (f16)
    ln_kernel<<<dim3(MROWS), 256, 0, stream>>>(
        x, ln2s + (size_t)l * DM, ln2b + (size_t)l * DM, xnh, nullptr, DM);
    // g = gelu(xn @ w1 + b1)  (N=4096) -> f16
    gemm(2, xnh, w1t + (size_t)l * FF * DM, b1 + (size_t)l * FF, nullptr, gh,
         MROWS, FF, DM);
    // x += g @ w2 + b2  (K=4096, N=1024)
    gemm(3, gh, w2t + (size_t)l * DM * FF, b2 + (size_t)l * DM, x, nullptr,
         MROWS, DM, FF);
  }

  // final LN -> d_out (fp32)
  ln_kernel<<<dim3(MROWS), 256, 0, stream>>>(x, lnfs, lnfb, nullptr,
                                             (float*)d_out, DM);
}
